// Llama3VisionCrossAttention_19859928777393
// MI455X (gfx1250) — matrix operand order repaired and checked by reading, not compiled, not run on hardware
//
#include <hip/hip_runtime.h>
#include <hip/hip_bf16.h>

typedef __attribute__((ext_vector_type(16))) __bf16 v16bf;
typedef __attribute__((ext_vector_type(8)))  float  v8f;
typedef unsigned short u16;
typedef unsigned int   u32;

// ---------------- problem constants ----------------
#define S_LEN    2048
#define KV_VALID 1601
#define KVP      1664      // KV padded to multiple of 64
#define D_MODEL  4096
#define H_ALL    32
#define HKV_N    8
#define HD_SZ    128
#define KV_HID   1024      // HKV_N * HD_SZ

// ---------------- CDNA5 async global->LDS path ----------------
#if __has_builtin(__builtin_amdgcn_global_load_async_to_lds_b128) && \
    __has_builtin(__builtin_amdgcn_s_wait_asynccnt)
#define HAVE_ASYNC_LDS 1
#else
#define HAVE_ASYNC_LDS 0
#endif

#if HAVE_ASYNC_LDS
// exact pointee type from the compiler diagnostic:
//   __attribute__((__vector_size__(4 * sizeof(int)))) int  (16-byte payload)
typedef __attribute__((__vector_size__(4 * sizeof(int)))) int v4i_pay;
typedef __attribute__((address_space(1))) v4i_pay* gptr_b128;
typedef __attribute__((address_space(3))) v4i_pay* lptr_b128;
#endif

__device__ __forceinline__ void async_copy16(const u16* g, u16* l) {
#if HAVE_ASYNC_LDS
  __builtin_amdgcn_global_load_async_to_lds_b128(
      (gptr_b128)(g), (lptr_b128)(l), /*offset=*/0, /*cpol=*/0);
#else
  *(uint4*)l = *(const uint4*)g;   // synchronous fallback
#endif
}

template <int N>
__device__ __forceinline__ void wait_async() {
#if HAVE_ASYNC_LDS
  __builtin_amdgcn_s_wait_asynccnt(N);
#endif
}

// ---------------- helpers ----------------
__device__ __forceinline__ u16 f2bf(float f) {
  u32 u = __float_as_uint(f);
  u32 r = u + 0x7FFFu + ((u >> 16) & 1u);   // round-to-nearest-even
  return (u16)(r >> 16);
}

union Frag { uint4 q[2]; v16bf v; };

// A-operand fragment (16x32 bf16), source row-major with K contiguous.
// lanes 0-15: row M=lane, K {0..7, 16..23}; lanes 16-31: row M=lane-16, K {8..15, 24..31}
__device__ __forceinline__ v16bf load_frag_a(const u16* base, int rowStride) {
  const int lane = threadIdx.x & 31;
  const int r  = lane & 15;
  const int hi = lane >> 4;
  const u16* p = base + r * rowStride + (hi ? 8 : 0);
  Frag f;
  f.q[0] = *(const uint4*)(p);
  f.q[1] = *(const uint4*)(p + 16);
  return f.v;
}

// B-operand fragment (32x16 bf16) from "Bt" storage (N-major, K contiguous).
// lanes 0-15: col N=lane, K {0..7, 16..23}; lanes 16-31: col N=lane-16, K {8..15, 24..31}
__device__ __forceinline__ v16bf load_frag_b(const u16* base, int rowStride) {
  const int lane = threadIdx.x & 31;
  const int n  = lane & 15;
  const int hi = lane >> 4;
  const u16* p = base + n * rowStride + (hi ? 8 : 0);
  Frag f;
  f.q[0] = *(const uint4*)(p);
  f.q[1] = *(const uint4*)(p + 16);
  return f.v;
}

__device__ __forceinline__ v8f wmma_bf16(v16bf a, v16bf b, v8f c) {
  return __builtin_amdgcn_wmma_f32_16x16x32_bf16(
      /*neg_a=*/false, a, /*neg_b=*/false, b,
      /*c_mod=*/(short)0, c, /*reuse_a=*/false, /*reuse_b=*/false);
}

// ---------------- kernel 1: fp32 -> bf16 transpose (weights) ----------------
// W: K x N fp32 (row major)  ->  WT: N x K bf16 (row major, K contiguous)
__global__ __launch_bounds__(256)
void transpose_bf16_kernel(const float* __restrict__ W, u16* __restrict__ WT,
                           int K, int N) {
  __shared__ float tile[32][33];
  const int n0 = blockIdx.x * 32;
  const int k0 = blockIdx.y * 32;
  const int tx = threadIdx.x & 31;
  const int ty = threadIdx.x >> 5;   // 0..7
#pragma unroll
  for (int i = 0; i < 4; ++i)
    tile[ty + i * 8][tx] = W[(size_t)(k0 + ty + i * 8) * N + n0 + tx];
  __syncthreads();
#pragma unroll
  for (int i = 0; i < 4; ++i)
    WT[(size_t)(n0 + ty + i * 8) * K + k0 + tx] = f2bf(tile[tx][ty + i * 8]);
}

// ---------------- kernel 2: RMSNorm -> bf16 (rows past validRows zero-filled) --
__global__ __launch_bounds__(256)
void rmsnorm_bf16_kernel(const float* __restrict__ x, const float* __restrict__ scale,
                         u16* __restrict__ out, int validRows, int D) {
  const int row = blockIdx.x;
  const int tid = threadIdx.x;
  u16* orow = out + (size_t)row * D;
  if (row >= validRows) {
    for (int i = tid; i < D / 8; i += 256)
      *(uint4*)&orow[i * 8] = make_uint4(0u, 0u, 0u, 0u);
    return;
  }
  const float* xr = x + (size_t)row * D;
  float ss = 0.f;
  for (int i = tid; i < D / 4; i += 256) {
    float4 v = ((const float4*)xr)[i];
    ss += v.x * v.x + v.y * v.y + v.z * v.z + v.w * v.w;
  }
#pragma unroll
  for (int m = 16; m >= 1; m >>= 1) ss += __shfl_xor(ss, m, 32);
  __shared__ float red[8];
  if ((tid & 31) == 0) red[tid >> 5] = ss;
  __syncthreads();
  float tot = 0.f;
#pragma unroll
  for (int i = 0; i < 8; ++i) tot += red[i];
  const float rs = rsqrtf(tot / (float)D + 1e-6f);
  for (int i = tid; i < D / 4; i += 256) {
    float4 v = ((const float4*)xr)[i];
    float4 s = ((const float4*)scale)[i];
    ushort4 o;
    o.x = f2bf(v.x * rs * s.x);
    o.y = f2bf(v.y * rs * s.y);
    o.z = f2bf(v.z * rs * s.z);
    o.w = f2bf(v.w * rs * s.w);
    *(ushort4*)&orow[i * 4] = o;
  }
}

// ---------------- kernel 3: WMMA GEMM  C = A(MxK) * Bt(NxK)^T --------------
// Double-buffered K-loop with async global->LDS prefetch (ASYNCcnt pipelining).
// OUT_MODE 0: store bf16.  OUT_MODE 1: fp32 out = residual + tanh(gate)*C.
#define A_STRIDE 40   // 32 + 8 u16 pad (80B rows, 16B aligned)

template <int OUT_MODE>
__global__ __launch_bounds__(256)
void gemm_wmma(const u16* __restrict__ A, const u16* __restrict__ Bt,
               void* __restrict__ Cout, int M, int N, int K, int ldc,
               const float* __restrict__ residual, const float* __restrict__ gate) {
  __shared__ u16 As[2][128 * A_STRIDE];
  __shared__ u16 Bs[2][128 * A_STRIDE];

  const int n0 = blockIdx.x * 128;
  const int m0 = blockIdx.y * 128;
  const int tid = threadIdx.x;
  const int wave = tid >> 5;
  const int wm = wave >> 2;   // 0..1 -> 64-row slab
  const int wn = wave & 3;    // 0..3 -> 32-col slab

  v8f acc[4][2];
#pragma unroll
  for (int i = 0; i < 4; ++i)
#pragma unroll
    for (int j = 0; j < 2; ++j) acc[i][j] = (v8f){0.f,0.f,0.f,0.f,0.f,0.f,0.f,0.f};

  // issue one 128x32 A tile + 128x32 Bt tile into LDS buffer `buf`
  // (4 async b128 instructions per thread -> ASYNCcnt depth 4 per wave)
  auto issue_tile = [&](int buf, int k0) {
#pragma unroll
    for (int i = 0; i < 2; ++i) {           // 512 16B-chunks / 256 threads
      int c = tid + i * 256;
      int row = c >> 2;
      int kc = (c & 3) * 8;
      async_copy16(&A[(size_t)(m0 + row) * K + k0 + kc],
                   &As[buf][row * A_STRIDE + kc]);
      async_copy16(&Bt[(size_t)(n0 + row) * K + k0 + kc],
                   &Bs[buf][row * A_STRIDE + kc]);
    }
  };

  const int nk = K >> 5;
  issue_tile(0, 0);
  for (int it = 0; it < nk; ++it) {
    const int buf = it & 1;
    if (it + 1 < nk) {
      issue_tile(buf ^ 1, (it + 1) << 5);   // prefetch next tile
      wait_async<4>();                      // leave the 4 prefetch ops in flight
    } else {
      wait_async<0>();
    }
    __syncthreads();                        // all waves' current tile resident

    v16bf af[4], bfr[2];
#pragma unroll
    for (int mf = 0; mf < 4; ++mf)
      af[mf] = load_frag_a(&As[buf][(wm * 64 + mf * 16) * A_STRIDE], A_STRIDE);
#pragma unroll
    for (int nf = 0; nf < 2; ++nf)
      bfr[nf] = load_frag_b(&Bs[buf][(wn * 32 + nf * 16) * A_STRIDE], A_STRIDE);
#pragma unroll
    for (int mf = 0; mf < 4; ++mf)
#pragma unroll
      for (int nf = 0; nf < 2; ++nf)
        acc[mf][nf] = wmma_bf16(af[mf], bfr[nf], acc[mf][nf]);
    __syncthreads();                        // buffer free before it is re-filled
  }

  const int lane = tid & 31;
  const int col = lane & 15;
  const int hi = lane >> 4;
  float g = 0.f;
  if (OUT_MODE == 1) g = tanhf(gate[0]);
#pragma unroll
  for (int mf = 0; mf < 4; ++mf)
#pragma unroll
    for (int nf = 0; nf < 2; ++nf)
#pragma unroll
      for (int r = 0; r < 8; ++r) {
        const int gm = m0 + wm * 64 + mf * 16 + r + hi * 8;
        const int gn = n0 + wn * 32 + nf * 16 + col;
        const size_t idx = (size_t)gm * ldc + gn;
        if (OUT_MODE == 0) {
          ((u16*)Cout)[idx] = f2bf(acc[mf][nf][r]);
        } else {
          ((float*)Cout)[idx] = residual[idx] + g * acc[mf][nf][r];
        }
      }
}

// ---------------- kernel 4: flash attention (WMMA, online softmax) ----------
#define KSTR 136   // 128 + 8 u16 pad
#define VSTR 72    // 64 + 8 u16 pad

__global__ __launch_bounds__(256)
void attn_wmma(const u16* __restrict__ Q, const u16* __restrict__ Kp,
               const u16* __restrict__ Vt, const unsigned char* __restrict__ vmask,
               u16* __restrict__ Out) {
  __shared__ u16 Ks[64 * KSTR];    // K tile:  64 kv x 128 hd (hd contiguous)
  __shared__ u16 Vs[128 * VSTR];   // Vt tile: 128 hd x 64 kv (kv contiguous)
  __shared__ u16 Ps[128 * VSTR];   // P tile:  128 q  x 64 kv

  const int h   = blockIdx.x >> 4;   // head 0..31
  const int qt  = blockIdx.x & 15;   // q tile of 128 rows
  const int hkv = h >> 2;            // GQA: 4 heads share one kv head
  const int tid = threadIdx.x;
  const int wave = tid >> 5;
  const int lane = tid & 31;
  const int lr = lane & 15;
  const int hi = lane >> 4;

  // Q fragments resident in registers (A layout, loaded straight from global)
  v16bf qa[4];
  {
    const int srow = qt * 128 + wave * 16 + lr;
    const u16* qp = Q + (size_t)srow * D_MODEL + h * HD_SZ;
#pragma unroll
    for (int kk = 0; kk < 4; ++kk) {
      Frag f;
      const u16* p = qp + kk * 32 + (hi ? 8 : 0);
      f.q[0] = *(const uint4*)(p);
      f.q[1] = *(const uint4*)(p + 16);
      qa[kk] = f.v;
    }
  }

  v8f of[8];
#pragma unroll
  for (int d = 0; d < 8; ++d) of[d] = (v8f){0.f,0.f,0.f,0.f,0.f,0.f,0.f,0.f};
  float mrow[8], lrow[8];
#pragma unroll
  for (int r = 0; r < 8; ++r) { mrow[r] = -__builtin_inff(); lrow[r] = 0.f; }

  const float sscale = 0.08838834764831845f;   // 1/sqrt(128)

  for (int kt = 0; kt < KVP / 64; ++kt) {
    // async-stage K and Vt tiles into LDS (8 async b128 per thread)
#pragma unroll
    for (int i = 0; i < 4; ++i) {              // K tile: 1024 chunks
      int c = tid + i * 256;
      int row = c >> 4;
      int cc = (c & 15) * 8;
      async_copy16(&Kp[(size_t)(kt * 64 + row) * KV_HID + hkv * HD_SZ + cc],
                   &Ks[row * KSTR + cc]);
    }
#pragma unroll
    for (int i = 0; i < 4; ++i) {              // Vt tile: 1024 chunks
      int c = tid + i * 256;
      int row = c >> 3;
      int cc = (c & 7) * 8;
      async_copy16(&Vt[(size_t)(hkv * HD_SZ + row) * KVP + kt * 64 + cc],
                   &Vs[row * VSTR + cc]);
    }
    wait_async<0>();
    __syncthreads();

    // S = Q * K^T  (per wave: 16 q-rows x 64 kv)
    v8f s[4];
#pragma unroll
    for (int nf = 0; nf < 4; ++nf) {
      v8f c = (v8f){0.f,0.f,0.f,0.f,0.f,0.f,0.f,0.f};
#pragma unroll
      for (int kk = 0; kk < 4; ++kk) {
        v16bf b = load_frag_b(&Ks[(nf * 16) * KSTR + kk * 32], KSTR);
        c = wmma_bf16(qa[kk], b, c);
      }
      s[nf] = c;
    }

    // scale + mask (padded kv and vision_mask)
#pragma unroll
    for (int nf = 0; nf < 4; ++nf) {
      const int n = kt * 64 + nf * 16 + lr;
      const bool ok = (n < KV_VALID) && (vmask[n] != 0);
#pragma unroll
      for (int r = 0; r < 8; ++r)
        s[nf][r] = ok ? s[nf][r] * sscale : -__builtin_inff();
    }

    // row max of this tile (reduce across the 16-lane half; rows ride VGPR idx)
    float tm[8];
#pragma unroll
    for (int r = 0; r < 8; ++r) {
      float v = s[0][r];
      v = fmaxf(v, s[1][r]); v = fmaxf(v, s[2][r]); v = fmaxf(v, s[3][r]);
      tm[r] = v;
    }
#pragma unroll
    for (int m = 8; m >= 1; m >>= 1)
#pragma unroll
      for (int r = 0; r < 8; ++r) tm[r] = fmaxf(tm[r], __shfl_xor(tm[r], m, 32));

    float corr[8];
#pragma unroll
    for (int r = 0; r < 8; ++r) {
      const float mn = fmaxf(mrow[r], tm[r]);
      corr[r] = __expf(mrow[r] - mn);
      mrow[r] = mn;
    }

    // P = exp(S - m); row sums; spill P to LDS in bf16 (A-layout source)
    float rsum[8];
#pragma unroll
    for (int r = 0; r < 8; ++r) rsum[r] = 0.f;
#pragma unroll
    for (int nf = 0; nf < 4; ++nf)
#pragma unroll
      for (int r = 0; r < 8; ++r) {
        const float p = __expf(s[nf][r] - mrow[r]);
        rsum[r] += p;
        Ps[(wave * 16 + r + hi * 8) * VSTR + nf * 16 + lr] = f2bf(p);
      }
#pragma unroll
    for (int m = 8; m >= 1; m >>= 1)
#pragma unroll
      for (int r = 0; r < 8; ++r) rsum[r] += __shfl_xor(rsum[r], m, 32);
#pragma unroll
    for (int r = 0; r < 8; ++r) lrow[r] = lrow[r] * corr[r] + rsum[r];

    // rescale accumulators
#pragma unroll
    for (int d = 0; d < 8; ++d)
#pragma unroll
      for (int r = 0; r < 8; ++r) of[d][r] *= corr[r];

    // O += P * V   (K-dim = kv tile of 64 -> two 32-chunks)
#pragma unroll
    for (int kk = 0; kk < 2; ++kk) {
      v16bf a = load_frag_a(&Ps[(wave * 16) * VSTR + kk * 32], VSTR);
#pragma unroll
      for (int d = 0; d < 8; ++d) {
        v16bf b = load_frag_b(&Vs[(d * 16) * VSTR + kk * 32], VSTR);
        of[d] = wmma_bf16(a, b, of[d]);
      }
    }
    __syncthreads();
  }

  // normalize + store bf16 (row-major s x D, hd contiguous per head)
  float rl[8];
#pragma unroll
  for (int r = 0; r < 8; ++r) rl[r] = 1.0f / lrow[r];
#pragma unroll
  for (int d = 0; d < 8; ++d)
#pragma unroll
    for (int r = 0; r < 8; ++r) {
      const int srow = qt * 128 + wave * 16 + r + hi * 8;
      const int col = h * HD_SZ + d * 16 + lr;
      Out[(size_t)srow * D_MODEL + col] = f2bf(of[d][r] * rl[r]);
    }
}

// ---------------- host side ----------------
extern "C" void kernel_launch(void* const* d_in, const int* in_sizes, int n_in,
                              void* d_out, int out_size, void* d_ws, size_t ws_size,
                              hipStream_t stream) {
  const float* hidden   = (const float*)d_in[0];
  const float* vision   = (const float*)d_in[1];
  const float* q_scale  = (const float*)d_in[2];
  const float* kv_scale = (const float*)d_in[3];
  const float* Wq = (const float*)d_in[4];
  const float* Wk = (const float*)d_in[5];
  const float* Wv = (const float*)d_in[6];
  const float* Wo = (const float*)d_in[7];
  const float* gate = (const float*)d_in[8];
  const unsigned char* vmask = (const unsigned char*)d_in[9];
  float* out = (float*)d_out;

  u16* ws = (u16*)d_ws;
  // workspace layout (u16 element offsets) — ~155 MB total, L2-resident on MI455X
  const size_t OFF_WqT = 0;
  const size_t OFF_WkT = OFF_WqT + (size_t)D_MODEL * D_MODEL;
  const size_t OFF_WvT = OFF_WkT + (size_t)KV_HID * D_MODEL;
  const size_t OFF_WoT = OFF_WvT + (size_t)KV_HID * D_MODEL;
  const size_t OFF_Hn  = OFF_WoT + (size_t)D_MODEL * D_MODEL;
  const size_t OFF_Vn  = OFF_Hn  + (size_t)S_LEN * D_MODEL;
  const size_t OFF_Q   = OFF_Vn  + (size_t)KVP * D_MODEL;
  const size_t OFF_K   = OFF_Q   + (size_t)S_LEN * D_MODEL;
  const size_t OFF_Vt  = OFF_K   + (size_t)KVP * KV_HID;
  const size_t OFF_A   = OFF_Vt  + (size_t)KV_HID * KVP;

  // 1) weights -> bf16, transposed to N-major/K-contiguous
  transpose_bf16_kernel<<<dim3(D_MODEL / 32, D_MODEL / 32), 256, 0, stream>>>(
      Wq, ws + OFF_WqT, D_MODEL, D_MODEL);
  transpose_bf16_kernel<<<dim3(KV_HID / 32, D_MODEL / 32), 256, 0, stream>>>(
      Wk, ws + OFF_WkT, D_MODEL, KV_HID);
  transpose_bf16_kernel<<<dim3(KV_HID / 32, D_MODEL / 32), 256, 0, stream>>>(
      Wv, ws + OFF_WvT, D_MODEL, KV_HID);
  transpose_bf16_kernel<<<dim3(D_MODEL / 32, D_MODEL / 32), 256, 0, stream>>>(
      Wo, ws + OFF_WoT, D_MODEL, D_MODEL);

  // 2) RMSNorm -> bf16 (vision padded to KVP rows with zeros)
  rmsnorm_bf16_kernel<<<S_LEN, 256, 0, stream>>>(hidden, q_scale, ws + OFF_Hn,
                                                 S_LEN, D_MODEL);
  rmsnorm_bf16_kernel<<<KVP, 256, 0, stream>>>(vision, kv_scale, ws + OFF_Vn,
                                               KV_VALID, D_MODEL);

  // 3) projections
  // Q = Hn (2048x4096) * WqT^T  -> bf16 2048x4096
  gemm_wmma<0><<<dim3(D_MODEL / 128, S_LEN / 128), 256, 0, stream>>>(
      ws + OFF_Hn, ws + OFF_WqT, (void*)(ws + OFF_Q),
      S_LEN, D_MODEL, D_MODEL, D_MODEL, nullptr, nullptr);
  // K = Vn (1664x4096) * WkT^T -> bf16 1664x1024 (kv-major, hd contiguous)
  gemm_wmma<0><<<dim3(KV_HID / 128, KVP / 128), 256, 0, stream>>>(
      ws + OFF_Vn, ws + OFF_WkT, (void*)(ws + OFF_K),
      KVP, KV_HID, D_MODEL, KV_HID, nullptr, nullptr);
  // Vt = WvT (1024x4096) * Vn^T -> bf16 1024x1664 (hd-major, kv contiguous)
  gemm_wmma<0><<<dim3(KVP / 128, KV_HID / 128), 256, 0, stream>>>(
      ws + OFF_WvT, ws + OFF_Vn, (void*)(ws + OFF_Vt),
      KV_HID, KVP, D_MODEL, KVP, nullptr, nullptr);

  // 4) flash attention: 32 heads x 16 q-tiles of 128 rows
  attn_wmma<<<H_ALL * (S_LEN / 128), 256, 0, stream>>>(
      ws + OFF_Q, ws + OFF_K, ws + OFF_Vt, vmask, ws + OFF_A);

  // 5) out = hidden + tanh(gate) * (attn @ Wo)
  gemm_wmma<1><<<dim3(D_MODEL / 128, S_LEN / 128), 256, 0, stream>>>(
      ws + OFF_A, ws + OFF_WoT, (void*)out,
      S_LEN, D_MODEL, D_MODEL, D_MODEL, hidden, gate);
  (void)in_sizes; (void)n_in; (void)out_size; (void)ws_size;
}
